// GCNLayer_16612933501110
// MI455X (gfx1250) — matrix operand order repaired and checked by reading, not compiled, not run on hardware
//
#include <hip/hip_runtime.h>
#include <hip/hip_bf16.h>

#define N_NODES 50000
#define N_EDGES 800000
#define FEATS   128

typedef __attribute__((ext_vector_type(16))) _Float16 v16h;
typedef __attribute__((ext_vector_type(8)))  float    v8f;

// ---------------------------------------------------------------------------
// 1) Zero the workspace (degrees + agg) -- contiguous float region.
// ---------------------------------------------------------------------------
__global__ void gcn_zero_kernel(float* ws, int total) {
    int stride = gridDim.x * blockDim.x;
    for (int i = blockIdx.x * blockDim.x + threadIdx.x; i < total; i += stride)
        ws[i] = 0.0f;
}

// ---------------------------------------------------------------------------
// 2) Degree accumulation: out_deg[src]+=1, in_deg[dst]+=1 (float atomics).
// ---------------------------------------------------------------------------
__global__ void gcn_degree_kernel(const int* __restrict__ src,
                                  const int* __restrict__ dst,
                                  float* __restrict__ out_deg,
                                  float* __restrict__ in_deg) {
    int e = blockIdx.x * blockDim.x + threadIdx.x;
    if (e >= N_EDGES) return;
    unsafeAtomicAdd(&out_deg[src[e]], 1.0f);
    unsafeAtomicAdd(&in_deg[dst[e]], 1.0f);
}

// ---------------------------------------------------------------------------
// 3) deg -> rsqrt(max(deg,1)) in place, over out_deg and in_deg together (2N).
// ---------------------------------------------------------------------------
__global__ void gcn_rsqrt_kernel(float* __restrict__ deg, int n) {
    int i = blockIdx.x * blockDim.x + threadIdx.x;
    if (i >= n) return;
    float d = deg[i];
    d = d < 1.0f ? 1.0f : d;
    deg[i] = rsqrtf(d);
}

// ---------------------------------------------------------------------------
// 4) Edge scatter: agg[dst] += feat[src] * (edge_w * out_rdeg[src])
//    32 threads per edge, 4 feats per thread (float4 gather + 4 f32 atomics).
// ---------------------------------------------------------------------------
__global__ void gcn_scatter_kernel(const float* __restrict__ feat,
                                   const float* __restrict__ edge_w,
                                   const int*   __restrict__ src,
                                   const int*   __restrict__ dst,
                                   const float* __restrict__ out_rdeg,
                                   float* __restrict__ agg) {
    int tid = blockIdx.x * blockDim.x + threadIdx.x;
    int e = tid >> 5;              // edge index
    int c = (tid & 31) << 2;       // feature chunk start (0,4,...,124)
    if (e >= N_EDGES) return;
    int s = src[e];
    int d = dst[e];
    float w = edge_w[e] * out_rdeg[s];
    const float4 f = *(const float4*)(feat + (long)s * FEATS + c);
    float* ap = agg + (long)d * FEATS + c;
    unsafeAtomicAdd(ap + 0, f.x * w);
    unsafeAtomicAdd(ap + 1, f.y * w);
    unsafeAtomicAdd(ap + 2, f.z * w);
    unsafeAtomicAdd(ap + 3, f.w * w);
}

// ---------------------------------------------------------------------------
// 5) WMMA GEMM: out = (agg @ W) * in_rdeg[:,None] + bias
//    Block = 256 threads = 8 waves; block b owns rows [16b,16b+16),
//    wave w owns cols [16w,16w+16). K loop: 4x v_wmma_f32_16x16x32_f16.
//    50000 % 16 == 0 -> no tails, EXEC all-1s at every WMMA.
// ---------------------------------------------------------------------------
__global__ __launch_bounds__(256) void gcn_gemm_kernel(
        const float* __restrict__ agg,
        const float* __restrict__ Wm,
        const float* __restrict__ bias,
        const float* __restrict__ in_rdeg,
        float* __restrict__ out) {
    const int lane    = threadIdx.x & 31;
    const int wave    = threadIdx.x >> 5;
    const int rowBase = blockIdx.x << 4;   // 16-row stripe
    const int colBase = wave << 4;         // 16-col tile per wave
    const int half    = lane >> 4;         // 0 or 1
    const int l15     = lane & 15;
    const int m       = rowBase + l15;     // A row held by this lane
    const int n       = colBase + l15;     // B/D col held by this lane

    v8f acc = {};
    #pragma unroll
    for (int k0 = 0; k0 < FEATS; k0 += 32) {
        // --- A fragment: row m, K = k0+half*8+[0..7] and k0+16+half*8+[0..7]
        const float* arow = agg + (long)m * FEATS + k0 + half * 8;
        float4 a0 = *(const float4*)(arow + 0);
        float4 a1 = *(const float4*)(arow + 4);
        float4 a2 = *(const float4*)(arow + 16);
        float4 a3 = *(const float4*)(arow + 20);
        v16h a;
        a[0]  = (_Float16)a0.x; a[1]  = (_Float16)a0.y;
        a[2]  = (_Float16)a0.z; a[3]  = (_Float16)a0.w;
        a[4]  = (_Float16)a1.x; a[5]  = (_Float16)a1.y;
        a[6]  = (_Float16)a1.z; a[7]  = (_Float16)a1.w;
        a[8]  = (_Float16)a2.x; a[9]  = (_Float16)a2.y;
        a[10] = (_Float16)a2.z; a[11] = (_Float16)a2.w;
        a[12] = (_Float16)a3.x; a[13] = (_Float16)a3.y;
        a[14] = (_Float16)a3.z; a[15] = (_Float16)a3.w;

        // --- B fragment: col n, K = k0+half*8+[0..7] and k0+16+half*8+[0..7] (coalesced across lanes)
        const float* wp = Wm + (long)(k0 + half * 8) * FEATS + n;
        v16h bb;
        #pragma unroll
        for (int i = 0; i < 16; ++i)
            bb[i] = (_Float16)wp[(long)(i < 8 ? i : i + 8) * FEATS];

        // D = A*B + C  (f32 accumulate)
        acc = __builtin_amdgcn_wmma_f32_16x16x32_f16(
                  /*neg_a=*/false, a, /*neg_b=*/false, bb,
                  /*c_mod=*/(short)0, acc, /*reuse_a=*/false, /*reuse_b=*/false);
    }

    const float bn = bias[n];
    #pragma unroll
    for (int r = 0; r < 8; ++r) {
        int mm = rowBase + r + half * 8;      // D row for VGPR r
        out[(long)mm * FEATS + n] = acc[r] * in_rdeg[mm] + bn;
    }
}

// ---------------------------------------------------------------------------
extern "C" void kernel_launch(void* const* d_in, const int* in_sizes, int n_in,
                              void* d_out, int out_size, void* d_ws, size_t ws_size,
                              hipStream_t stream) {
    const float* feat    = (const float*)d_in[0];   // [N, 128]
    const float* Wm      = (const float*)d_in[1];   // [128, 128]
    const float* bias    = (const float*)d_in[2];   // [128]
    const float* edge_w  = (const float*)d_in[3];   // [E]
    const int*   e_src   = (const int*)d_in[4];     // [E]
    const int*   e_dst   = (const int*)d_in[5];     // [E]
    float*       out     = (float*)d_out;           // [N, 128]

    // Workspace layout (floats): out_rdeg[N] | in_rdeg[N] | agg[N*128]
    float* out_rdeg = (float*)d_ws;
    float* in_rdeg  = out_rdeg + N_NODES;
    float* agg      = in_rdeg + N_NODES;
    const int ws_floats = 2 * N_NODES + N_NODES * FEATS;

    // 1) zero degrees + agg
    gcn_zero_kernel<<<4096, 256, 0, stream>>>((float*)d_ws, ws_floats);

    // 2) degrees
    gcn_degree_kernel<<<(N_EDGES + 255) / 256, 256, 0, stream>>>(
        e_src, e_dst, out_rdeg, in_rdeg);

    // 3) rsqrt(max(deg,1)) on both arrays (contiguous 2N)
    gcn_rsqrt_kernel<<<(2 * N_NODES + 255) / 256, 256, 0, stream>>>(
        out_rdeg, 2 * N_NODES);

    // 4) scatter-aggregate: 32 threads/edge
    {
        long threads = (long)N_EDGES * 32;
        int blocks = (int)((threads + 255) / 256);
        gcn_scatter_kernel<<<blocks, 256, 0, stream>>>(
            feat, edge_w, e_src, e_dst, out_rdeg, agg);
    }

    // 5) WMMA GEMM + dst-norm + bias (N_NODES/16 = 3125 blocks, 8 waves each)
    gcn_gemm_kernel<<<N_NODES / 16, 256, 0, stream>>>(
        agg, Wm, bias, in_rdeg, out);
}
